// CVAEEncoder_11751030522624
// MI455X (gfx1250) — matrix operand order repaired and checked by reading, not compiled, not run on hardware
//
#include <hip/hip_runtime.h>

// Problem constants (match reference)
#define Bsz  64
#define CIN  64
#define NN   4097      // nodes incl. null node 0
#define TT   4096      // real nodes
#define HIDD 128
#define LATD 64
#define EPSV 1e-5f

typedef __attribute__((ext_vector_type(16))) __bf16 v16bf;
typedef __attribute__((ext_vector_type(8)))  float  v8f;

// ---- bf16 <-> f32 without __bf16 scalar arithmetic (RNE) ----
__device__ __forceinline__ unsigned short f2bf(float x) {
  unsigned u = __float_as_uint(x);
  u += 0x7FFFu + ((u >> 16) & 1u);
  return (unsigned short)(u >> 16);
}
__device__ __forceinline__ float bf2f(unsigned short h) {
  return __uint_as_float(((unsigned)h) << 16);
}

union FragAB { v16bf v; uint4 u[2]; };

// ---------------- init: zero layer-norm stats -----------------
__global__ void init_stats(float* s, int n) {
  for (int i = threadIdx.x; i < n; i += blockDim.x) s[i] = 0.f;
}

// ---------------- fused gather + GEMM (WMMA) ------------------
// One block: 64 output nodes x 128 output channels for one batch b.
// D(128 x 64) = W(128 x 3C) * G(3C x 64), G gathered via children.
// Cin and input dtype are compile-time -> K loop fully unrolls.
// Writes bf16 conv output (node column t -> slab column t+1) and
// accumulates exact f32 sum / sum-of-squares for the layer norm.
template <int Cin, bool IN_BF16>
__global__ void __launch_bounds__(256)
tree_conv_wmma(const void*  __restrict__ in,        // f32 slab or bf16 slab
               const float* __restrict__ W,         // (HIDD, Cin, 3) f32
               const float* __restrict__ bias,      // (HIDD,) f32
               const int*   __restrict__ children,  // (Bsz, 3*TT)
               unsigned short* __restrict__ out,    // (Bsz, HIDD, NN) bf16
               float* __restrict__ stats)           // (Bsz, 2): sum, sumsq
{
  constexpr int K3C = 3 * Cin;
  const int b   = blockIdx.y;
  const int t0  = blockIdx.x * 64;            // first real node of tile
  const int tid = threadIdx.x;
  const int lane = tid & 31;
  const int wave = tid >> 5;                  // 8 waves -> 8 M-bands of 16
  const int half = lane >> 4;                 // lane group (ISA frag layout)
  const int mrow = lane & 15;

  const float*          inF = (const float*)in;
  const unsigned short* inB = (const unsigned short*)in;

  __shared__ __align__(16) unsigned short As[HIDD * 32]; // W K-slice, row-major stride 32
  __shared__ __align__(16) unsigned short Bs[64 * 32];   // gathered G slice, [node][k]
  __shared__ int   childs[192];
  __shared__ float s_sum, s_sq;

  if (tid < 192) childs[tid] = children[b * 3 * TT + t0 * 3 + tid];

  v8f acc[4];
  #pragma unroll
  for (int j = 0; j < 4; ++j)
    #pragma unroll
    for (int r = 0; r < 8; ++r) acc[j][r] = 0.f;

  #pragma unroll
  for (int kk0 = 0; kk0 < K3C; kk0 += 32) {
    __syncthreads();   // prior-iter frag reads done; childs visible on iter 0

    if (kk0 + 32 < K3C)                       // gfx1250 global_prefetch_b8
      __builtin_prefetch(&W[(tid >> 1) * K3C + kk0 + 32], 0, 0);

    // Stage A = W[:, kk0:kk0+32] (f32 -> bf16). 4096 elems / 256 thr.
    #pragma unroll
    for (int i = 0; i < 16; ++i) {
      int idx = tid + i * 256;
      int o = idx >> 5, kl = idx & 31;
      As[idx] = f2bf(W[o * K3C + kk0 + kl]);
    }
    // Stage B = gathered G[kk0:kk0+32, t0:t0+64]. 2048 elems / 256 thr.
    #pragma unroll
    for (int i = 0; i < 8; ++i) {
      int idx  = tid + i * 256;
      int node = idx >> 5, kl = idx & 31;
      int kk = kk0 + kl;
      int c  = kk / 3, r = kk - 3 * c;
      int ch = childs[node * 3 + r];
      int off = (b * Cin + c) * NN + ch;
      float v = IN_BF16 ? bf2f(inB[off]) : inF[off];
      Bs[idx] = f2bf(v);
    }
    __syncthreads();

    // A fragment: band row = mrow; K runs {half*8..+7} and {16+half*8..+7}
    FragAB a;
    a.u[0] = *(const uint4*)&As[(wave * 16 + mrow) * 32 + half * 8];
    a.u[1] = *(const uint4*)&As[(wave * 16 + mrow) * 32 + 16 + half * 8];
    #pragma unroll
    for (int j = 0; j < 4; ++j) {
      FragAB bb;                               // B col = j*16+mrow, K runs {half*8..+7} and {16+half*8..+7}
      bb.u[0] = *(const uint4*)&Bs[(j * 16 + mrow) * 32 + half * 8];
      bb.u[1] = *(const uint4*)&Bs[(j * 16 + mrow) * 32 + 16 + half * 8];
      acc[j] = __builtin_amdgcn_wmma_f32_16x16x32_bf16(
          false, a.v, false, bb.v, (short)0, acc[j], false, false);
    }
  }

  // Bias, exact LN partial stats, bf16 store (slab column = node+1).
  float bv[8];
  #pragma unroll
  for (int r = 0; r < 8; ++r) bv[r] = bias[wave * 16 + half * 8 + r];

  float lsum = 0.f, lsq = 0.f;
  #pragma unroll
  for (int j = 0; j < 4; ++j) {
    int node = 1 + t0 + j * 16 + mrow;
    #pragma unroll
    for (int r = 0; r < 8; ++r) {
      int o = wave * 16 + half * 8 + r;
      float val = acc[j][r] + bv[r];
      lsum += val; lsq += val * val;
      out[(b * HIDD + o) * NN + node] = f2bf(val);
    }
  }
  if (tid == 0) { s_sum = 0.f; s_sq = 0.f; }
  __syncthreads();
  atomicAdd(&s_sum, lsum);                    // ds_add_f32
  atomicAdd(&s_sq,  lsq);
  __syncthreads();
  if (tid == 0) {
    atomicAdd(&stats[b * 2 + 0], s_sum);      // global_atomic_add_f32
    atomicAdd(&stats[b * 2 + 1], s_sq);
  }
}

// ------------- tree layer norm (ddof=1) + ReLU, in place -------------
// Null column (node 0) is logically zero on input; we write its
// normalized value so later gathers (children may index 0) see it.
__global__ void __launch_bounds__(256)
norm_relu(unsigned short* __restrict__ x, const float* __restrict__ stats)
{
  long idx = (long)blockIdx.x * 256 + threadIdx.x;
  const long per = (long)HIDD * NN;
  if (idx >= (long)Bsz * per) return;
  int  b    = (int)(idx / per);
  long rem  = idx - (long)b * per;
  int  node = (int)(rem % NN);

  const float n    = (float)(HIDD * NN);
  float sum  = stats[b * 2 + 0];
  float sq   = stats[b * 2 + 1];
  float mean = sum / n;
  float var  = (sq - sum * sum / n) / (n - 1.f);
  float inv  = 1.f / (sqrtf(var) + EPSV);

  float v = (node == 0) ? 0.f : bf2f(x[idx]);
  float y = (v - mean) * inv;
  x[idx] = f2bf(fmaxf(y, 0.f));
}

// ---------------- dynamic pooling: max over nodes ----------------
__global__ void __launch_bounds__(256)
maxpool(const unsigned short* __restrict__ x, float* __restrict__ embed, int col_off)
{
  int row = blockIdx.x;                       // b*HIDD + o
  int b = row / HIDD, o = row - b * HIDD;
  const unsigned short* p = x + (long)row * NN;
  float m = -3.4e38f;
  for (int i = threadIdx.x; i < NN; i += 256) m = fmaxf(m, bf2f(p[i]));
  __shared__ float red[256];
  red[threadIdx.x] = m;
  __syncthreads();
  for (int s = 128; s > 0; s >>= 1) {
    if (threadIdx.x < s) red[threadIdx.x] = fmaxf(red[threadIdx.x], red[threadIdx.x + s]);
    __syncthreads();
  }
  if (threadIdx.x == 0) embed[b * 256 + col_off + o] = red[0];
}

// ---------------- FC heads: mu, logvar ----------------
__global__ void __launch_bounds__(256)
fc_head(const float* __restrict__ embed,
        const float* __restrict__ Wmu, const float* __restrict__ bmu,
        const float* __restrict__ Wlv, const float* __restrict__ blv,
        float* __restrict__ outp)
{
  int tid = blockIdx.x * 256 + threadIdx.x;
  if (tid >= 2 * Bsz * LATD) return;
  int which = tid / (Bsz * LATD);
  int rem   = tid - which * (Bsz * LATD);
  int b = rem / LATD, j = rem - b * LATD;
  const float* Wp = which ? Wlv : Wmu;
  const float* bp = which ? blv : bmu;
  float a = bp[j];
  const float* e = embed + b * 256;
  const float* w = Wp + j * 256;
  #pragma unroll 8
  for (int i = 0; i < 256; ++i) a += e[i] * w[i];
  outp[which * Bsz * LATD + b * LATD + j] = a;
}

// ======================= launcher =======================
extern "C" void kernel_launch(void* const* d_in, const int* in_sizes, int n_in,
                              void* d_out, int out_size, void* d_ws, size_t ws_size,
                              hipStream_t stream)
{
  (void)in_sizes; (void)n_in; (void)out_size; (void)ws_size;
  const float* lf  = (const float*)d_in[0];
  const float* pf  = (const float*)d_in[1];
  const int*   ch  = (const int*)  d_in[2];
  const float* Wl1 = (const float*)d_in[3];  const float* bl1 = (const float*)d_in[4];
  const float* Wl2 = (const float*)d_in[5];  const float* bl2 = (const float*)d_in[6];
  const float* Wp1 = (const float*)d_in[7];  const float* bp1 = (const float*)d_in[8];
  const float* Wp2 = (const float*)d_in[9];  const float* bp2 = (const float*)d_in[10];
  const float* Wmu = (const float*)d_in[11]; const float* bmu = (const float*)d_in[12];
  const float* Wlv = (const float*)d_in[13]; const float* blv = (const float*)d_in[14];

  char* ws = (char*)d_ws;
  const size_t xbytes = (size_t)Bsz * HIDD * NN * sizeof(unsigned short); // 67 MB
  unsigned short* x1 = (unsigned short*)ws;  ws += xbytes;
  unsigned short* x2 = (unsigned short*)ws;  ws += xbytes;
  float* embed = (float*)ws;                 ws += (size_t)Bsz * 256 * sizeof(float);
  float* stats = (float*)ws;                 // 4 phases * Bsz * 2 floats

  init_stats<<<1, 256, 0, stream>>>(stats, 4 * Bsz * 2);

  dim3 cgrid(TT / 64, Bsz);
  const long nElem  = (long)Bsz * HIDD * NN;
  const int  nblk   = (int)((nElem + 255) / 256);

  // ---- logical encoder ----
  tree_conv_wmma<CIN,  false><<<cgrid, 256, 0, stream>>>(lf, Wl1, bl1, ch, x1, stats + 0 * Bsz * 2);
  norm_relu<<<nblk, 256, 0, stream>>>(x1, stats + 0 * Bsz * 2);
  tree_conv_wmma<HIDD, true ><<<cgrid, 256, 0, stream>>>(x1, Wl2, bl2, ch, x2, stats + 1 * Bsz * 2);
  norm_relu<<<nblk, 256, 0, stream>>>(x2, stats + 1 * Bsz * 2);
  maxpool<<<Bsz * HIDD, 256, 0, stream>>>(x2, embed, 0);

  // ---- physical encoder (same children, per reference) ----
  tree_conv_wmma<CIN,  false><<<cgrid, 256, 0, stream>>>(pf, Wp1, bp1, ch, x1, stats + 2 * Bsz * 2);
  norm_relu<<<nblk, 256, 0, stream>>>(x1, stats + 2 * Bsz * 2);
  tree_conv_wmma<HIDD, true ><<<cgrid, 256, 0, stream>>>(x1, Wp2, bp2, ch, x2, stats + 3 * Bsz * 2);
  norm_relu<<<nblk, 256, 0, stream>>>(x2, stats + 3 * Bsz * 2);
  maxpool<<<Bsz * HIDD, 256, 0, stream>>>(x2, embed, HIDD);

  fc_head<<<(2 * Bsz * LATD + 255) / 256, 256, 0, stream>>>(embed, Wmu, bmu, Wlv, blv, (float*)d_out);
}
